// LatentTSDCD_19447611916909
// MI455X (gfx1250) — compile-verified, matrix operand order checked by reading, not run on hardware
//
#include <hip/hip_runtime.h>
#include <math.h>

#define DEV __device__ __forceinline__

typedef __attribute__((ext_vector_type(16))) _Float16 v16h;
typedef __attribute__((ext_vector_type(8)))  _Float16 v8h;
typedef __attribute__((ext_vector_type(8)))  float    v8f;
typedef __attribute__((ext_vector_type(4)))  unsigned u32x4;
typedef __attribute__((ext_vector_type(8)))  int      i32x8;
typedef __attribute__((ext_vector_type(4)))  int      i32x4;

constexpr int Bc = 512, TAUc = 4, Dc = 5, DXc = 256, DZc = 16, Hc = 64;
constexpr int Mc  = Dc * DZc;     // 80
constexpr int TMc = TAUc * Mc;    // 320
constexpr int Rp  = Bc * TAUc;    // past rows per d = 2048
constexpr int Rc  = Rp + Bc;      // rows per d = 2560 (past + current)
constexpr float kLOG2PI = 1.8378770664093453f;

// ---------------- workspace layout (bytes) ----------------
constexpr size_t alup(size_t x) { return (x + 255) & ~(size_t)255; }
constexpr size_t OFF_ACC  = 0;                                   // accum[0]=recon, accum[1]=kl
constexpr size_t OFF_AENC = 256;                                 // f16 [d][r][x]   (x/y rows grouped by d)
constexpr size_t SZ_AENC  = (size_t)Dc * Rc * DXc * 2;
constexpr size_t OFF_W0E  = alup(OFF_AENC + SZ_AENC);            // f16 [d][n=z*64+h][k=x]  (w_enc folded in)
constexpr size_t SZ_W0E   = (size_t)Dc * DZc * Hc * DXc * 2;
constexpr size_t OFF_W0D  = alup(OFF_W0E + SZ_W0E);              // f16 [i][n=j*64+h][k=32] (w_dec folded, K padded)
constexpr size_t SZ_W0D   = (size_t)Dc * DXc * Hc * 32 * 2;
constexpr size_t OFF_TW0  = alup(OFF_W0D + SZ_W0D);              // f16 [m][n=h][k=i]
constexpr size_t SZ_TW0   = (size_t)Mc * Hc * TMc * 2;
constexpr size_t OFF_ATR  = alup(OFF_TW0 + SZ_TW0);              // f16 [m][b][i]  adjacency-gated past latents
constexpr size_t SZ_ATR   = (size_t)Mc * Bc * TMc * 2;
constexpr size_t OFF_ZP   = alup(OFF_ATR + SZ_ATR);              // f32 z_past (b,tau,M)
constexpr size_t SZ_ZP    = (size_t)Bc * TAUc * Mc * 4;
constexpr size_t OFF_MUT  = alup(OFF_ZP + SZ_ZP);                // f32 mu_t (b,D,DZ)
constexpr size_t SZ_MUT   = (size_t)Bc * Dc * DZc * 4;
constexpr size_t OFF_ZT   = alup(OFF_MUT + SZ_MUT);              // f16 z_t [i][b][32] (K padded)
constexpr size_t SZ_ZT    = (size_t)Dc * Bc * 32 * 2;
constexpr size_t OFF_PZ   = alup(OFF_ZT + SZ_ZT);                // f32 pz_mu (b,M)
constexpr size_t SZ_PZ    = (size_t)Bc * Mc * 4;

// ---------------- helpers ----------------
DEV unsigned pcg(unsigned v) {
  v = v * 747796405u + 2891336453u;
  unsigned w = ((v >> ((v >> 28) + 4u)) ^ v) * 277803737u;
  return (w >> 22) ^ w;
}
DEV float u01(unsigned s) { return ((pcg(s) >> 8) + 0.5f) * (1.0f / 16777216.0f); }
DEV float gaussf(unsigned s) {
  float u1 = u01(s * 2u + 1u), u2 = u01(s * 2u + 2u);
  return sqrtf(-2.0f * __logf(u1)) * __cosf(6.283185307179586f * u2);
}

// 16x32 f16 fragment load (A, or B from an N-major weight panel).
// lanes 0-15: row=lane, halves K {0..7,16..23}; lanes 16-31: row=lane-16, K {8..15,24..31}.
DEV v16h ldfrag(const _Float16* __restrict__ base, int ld, int lane) {
  const int r = lane & 15, sel = (lane >> 4) & 1;
  const _Float16* p = base + (size_t)r * ld + sel * 8;
  v8h lo = *(const v8h*)(p);
  v8h hi = *(const v8h*)(p + 16);
  return __builtin_shufflevector(lo, hi, 0,1,2,3,4,5,6,7,8,9,10,11,12,13,14,15);
}

DEV v8f wmma32(v16h a, v16h b, v8f c) {
  return __builtin_amdgcn_wmma_f32_16x16x32_f16(false, a, false, b, (short)0, c, false, false);
}

// Tensor Data Mover: DMA a contiguous 2D panel (rows x k_elems f16, row stride
// == k_elems) from global memory into LDS. D# per CDNA5 ISA:
// group0 = {count=1 | lds_addr | global_addr[56:0] | type=2},
// group1 = {data_size=2B, tensor_dim0=K, tensor_dim1=rows, tile_dim0=K,
//           tile_dim1=rows, tensor_dim0_stride=K}; groups 2/3 zero (<=2D tensor).
// NOTE: this toolchain exposes the 6-arg builtin (clang-23 / therock-10.0):
//   (u32x4 g0, i32x8 g1, i32x4, i32x4, i32x8, i32 cpol)
DEV void tdm_load_panel(unsigned lds_byte_off, const _Float16* gsrc,
                        unsigned k_elems, unsigned rows) {
  unsigned long long ga = (unsigned long long)gsrc;
  u32x4 g0;
  g0[0] = 1u;                                                  // count=1, user descriptor
  g0[1] = lds_byte_off;                                        // lds_addr
  g0[2] = (unsigned)ga;                                        // global_addr[31:0]
  g0[3] = (unsigned)((ga >> 32) & 0x1ffffffu) | (2u << 30);    // global_addr[56:32] | type=2
  i32x8 g1;
  g1[0] = 0x00010000;                                          // data_size=1 (2 bytes)
  g1[1] = (int)((k_elems & 0xffffu) << 16);                    // tensor_dim0[15:0]
  g1[2] = (int)(((k_elems >> 16) & 0xffffu) | ((rows & 0xffffu) << 16)); // td0 hi | td1 lo
  g1[3] = (int)(((rows >> 16) & 0xffffu) | ((k_elems & 0xffffu) << 16)); // td1 hi | tile_dim0
  g1[4] = (int)(rows & 0xffffu);                               // tile_dim1 (tile_dim2=0)
  g1[5] = (int)k_elems;                                        // tensor_dim0_stride[31:0]
  g1[6] = 0;
  g1[7] = 0;
  i32x4 zz4 = {0, 0, 0, 0};
  i32x8 zz8 = {0, 0, 0, 0, 0, 0, 0, 0};
  __builtin_amdgcn_tensor_load_to_lds(g0, g1, zz4, zz4, zz8, 0);
}

DEV unsigned lds_off_of(const void* p) {    // flat LDS aperture: offset = addr[31:0]
  return (unsigned)(unsigned long long)p;
}

// ---------------- prep kernels ----------------
__global__ __launch_bounds__(256) void k_init(float* acc, _Float16* zt16) {
  int idx = blockIdx.x * 256 + threadIdx.x;
  if (idx < Dc * Bc * 32) zt16[idx] = (_Float16)0.0f;
  if (idx < 8) acc[idx] = 0.0f;
}

__global__ __launch_bounds__(256) void k_build_aenc(const float* __restrict__ x,
                                                    const float* __restrict__ y,
                                                    _Float16* __restrict__ Aenc) {
  int idx = blockIdx.x * 256 + threadIdx.x;
  if (idx >= Dc * Rc * DXc) return;
  int k = idx % DXc;
  int r = (idx / DXc) % Rc;
  int d = idx / (DXc * Rc);
  float v;
  if (r < Rp) {
    int b = r / TAUc, tau = r % TAUc;
    v = x[(((size_t)(b * TAUc + tau)) * Dc + d) * DXc + k];
  } else {
    int b = r - Rp;
    v = y[((size_t)(b * Dc + d)) * DXc + k];
  }
  Aenc[idx] = (_Float16)v;
}

__global__ __launch_bounds__(256) void k_fold_enc(const float* __restrict__ w_enc,
                                                  const float* __restrict__ enc_W0,
                                                  _Float16* __restrict__ W0e) {
  int idx = blockIdx.x * 256 + threadIdx.x;
  if (idx >= Dc * DZc * Hc * DXc) return;
  int k = idx % DXc;
  int n = (idx / DXc) % (DZc * Hc);
  int d = idx / (DXc * DZc * Hc);
  int z = n >> 6, h = n & 63;
  float v = w_enc[(size_t)(d * DZc + z) * DXc + k] * enc_W0[((size_t)z * DXc + k) * Hc + h];
  W0e[idx] = (_Float16)v;
}

__global__ __launch_bounds__(256) void k_fold_dec(const float* __restrict__ w_dec,
                                                  const float* __restrict__ dec_W0,
                                                  _Float16* __restrict__ W0d) {
  int idx = blockIdx.x * 256 + threadIdx.x;
  if (idx >= Dc * DXc * Hc * 32) return;
  int k = idx & 31;
  int n = (idx >> 5) % (DXc * Hc);
  int i = idx / (32 * DXc * Hc);
  int j = n >> 6, h = n & 63;
  float v = 0.0f;
  if (k < DZc)
    v = w_dec[((size_t)(i * DXc + j)) * DZc + k] * dec_W0[((size_t)j * DZc + k) * Hc + h];
  W0d[idx] = (_Float16)v;
}

__global__ __launch_bounds__(256) void k_fold_t(const float* __restrict__ t_W0,
                                                _Float16* __restrict__ tW0h) {
  int idx = blockIdx.x * 256 + threadIdx.x;
  if (idx >= Mc * Hc * TMc) return;
  int k = idx % TMc;
  int h = (idx / TMc) % Hc;
  int m = idx / (TMc * Hc);
  tW0h[idx] = (_Float16)t_W0[((size_t)m * TMc + k) * Hc + h];
}

// ---------------- encoder: TDM-staged weight panel + WMMA + epilogue ----------------
__global__ __launch_bounds__(256) void k_encode(const _Float16* __restrict__ Aenc,
                                                const _Float16* __restrict__ W0e,
                                                const float* __restrict__ enc_b0,
                                                const float* __restrict__ enc_W1,
                                                const float* __restrict__ enc_b1,
                                                const float* __restrict__ lv_enc,
                                                float* __restrict__ zp,
                                                float* __restrict__ mu_t,
                                                _Float16* __restrict__ zt16) {
  __shared__ _Float16 Bs[Hc * DXc];                 // 32 KB weight panel for (d, zi)
  const int wg  = blockIdx.x;                       // 1600 = Dc * DZc * 20
  const int rtg = wg % 20;
  const int zi  = (wg / 20) % DZc;
  const int d   = wg / (20 * DZc);
  const _Float16* Wb = W0e + ((size_t)d * DZc + zi) * (size_t)(Hc * DXc);
  if ((threadIdx.x >> 5) == 0) {                    // one TDM op per workgroup
    tdm_load_panel(lds_off_of(Bs), Wb, DXc, Hc);
    __builtin_amdgcn_s_wait_tensorcnt(0);
  }
  __syncthreads();

  const int lane = threadIdx.x & 31;
  const int wv   = threadIdx.x >> 5;                // 0..7
  const int rt   = rtg * 8 + wv;                    // 0..159
  const _Float16* Ab = Aenc + ((size_t)d * Rc + rt * 16) * DXc;
  const v8f z8 = {0.f,0.f,0.f,0.f,0.f,0.f,0.f,0.f};
  v8f acc[4] = {z8, z8, z8, z8};
  for (int kc = 0; kc < DXc; kc += 32) {
    v16h a = ldfrag(Ab + kc, DXc, lane);
    if (kc + 32 < DXc) __builtin_prefetch(Ab + kc + 32, 0, 0);
#pragma unroll
    for (int t = 0; t < 4; ++t) {
      v16h b = ldfrag(&Bs[(t * 16) * DXc + kc], DXc, lane);
      acc[t] = wmma32(a, b, acc[t]);
    }
  }
  const int col = lane & 15;
  float w1v[4], b0v[4];
#pragma unroll
  for (int t = 0; t < 4; ++t) {
    int h = t * 16 + col;
    w1v[t] = enc_W1[zi * Hc + h];
    b0v[t] = enc_b0[zi * Hc + h];
  }
  float pv[8];
#pragma unroll
  for (int v = 0; v < 8; ++v) {
    float s = 0.f;
#pragma unroll
    for (int t = 0; t < 4; ++t) {
      float hv = acc[t][v] + b0v[t];
      hv = hv > 0.f ? hv : 0.01f * hv;              // leaky_relu(0.01)
      s += hv * w1v[t];
    }
    pv[v] = s;
  }
#pragma unroll
  for (int off = 1; off < 16; off <<= 1)
#pragma unroll
    for (int v = 0; v < 8; ++v) pv[v] += __shfl_xor(pv[v], off, 32);
  if (lane == 0 || lane == 16) {
    const int roff = (lane == 16) ? 8 : 0;
    const float qstd = __expf(0.5f * lv_enc[d]);
#pragma unroll
    for (int v = 0; v < 8; ++v) {
      int r = rt * 16 + v + roff;
      float mu = pv[v] + enc_b1[zi];
      if (r < Rp) {
        int b = r / TAUc, tau = r % TAUc;
        unsigned g = (unsigned)((((b * TAUc) + tau) * Dc + d) * DZc + zi);
        zp[((b * TAUc) + tau) * Mc + d * DZc + zi] = mu + qstd * gaussf(g ^ 0x51ed270bu);
      } else {
        int b = r - Rp;
        unsigned g = (unsigned)(((b * Dc) + d) * DZc + zi);
        mu_t[((b * Dc) + d) * DZc + zi] = mu;
        zt16[((size_t)d * Bc + b) * 32 + zi] = (_Float16)(mu + qstd * gaussf(g ^ 0x9e3779b9u));
      }
    }
  }
}

// ---------------- adjacency-gated past latents (transition A-matrix) ----------------
__global__ __launch_bounds__(256) void k_build_atr(const float* __restrict__ mask_param,
                                                   const float* __restrict__ zp,
                                                   _Float16* __restrict__ Atr) {
  long long idx = (long long)blockIdx.x * 256 + threadIdx.x;
  if (idx >= (long long)Mc * Bc * TMc) return;
  int i = (int)(idx % TMc);
  int b = (int)((idx / TMc) % Bc);
  int m = (int)(idx / ((long long)TMc * Bc));
  int tau = i / Mc, src = i % Mc;
  // gumbel-sigmoid soft sample (temperature 1)
  float u = u01((unsigned)((((b * TAUc + tau) * Mc) + m) * Mc + src) ^ 0x2545f491u);
  float logit = __logf(u) - log1pf(-u);
  float a = 1.0f / (1.0f + __expf(-(mask_param[((size_t)tau * Mc + m) * Mc + src] + logit)));
  Atr[idx] = (_Float16)(a * zp[((b * TAUc) + tau) * Mc + src]);
}

// ---------------- transition: TDM-staged per-node panel + WMMA ----------------
__global__ __launch_bounds__(256) void k_trans(const _Float16* __restrict__ Atr,
                                               const _Float16* __restrict__ tW0h,
                                               const float* __restrict__ t_b0,
                                               const float* __restrict__ t_W1,
                                               const float* __restrict__ t_b1,
                                               float* __restrict__ pz_mu) {
  __shared__ _Float16 Bs[Hc * TMc];                 // 40 KB weight panel for node m
  const int wg  = blockIdx.x;                       // 320 = Mc * 4
  const int rtg = wg & 3;
  const int m   = wg >> 2;
  const _Float16* Wb = tW0h + (size_t)m * Hc * TMc;
  if ((threadIdx.x >> 5) == 0) {
    tdm_load_panel(lds_off_of(Bs), Wb, TMc, Hc);
    __builtin_amdgcn_s_wait_tensorcnt(0);
  }
  __syncthreads();

  const int lane = threadIdx.x & 31;
  const int wv   = threadIdx.x >> 5;
  const int rt   = rtg * 8 + wv;                    // 0..31
  const _Float16* Ab = Atr + ((size_t)m * Bc + rt * 16) * TMc;
  const v8f z8 = {0.f,0.f,0.f,0.f,0.f,0.f,0.f,0.f};
  v8f acc[4] = {z8, z8, z8, z8};
  for (int kc = 0; kc < TMc; kc += 32) {
    v16h a = ldfrag(Ab + kc, TMc, lane);
    if (kc + 32 < TMc) __builtin_prefetch(Ab + kc + 32, 0, 0);
#pragma unroll
    for (int t = 0; t < 4; ++t) {
      v16h b = ldfrag(&Bs[(t * 16) * TMc + kc], TMc, lane);
      acc[t] = wmma32(a, b, acc[t]);
    }
  }
  const int col = lane & 15;
  float w1v[4], b0v[4];
#pragma unroll
  for (int t = 0; t < 4; ++t) {
    int h = t * 16 + col;
    w1v[t] = t_W1[m * Hc + h];
    b0v[t] = t_b0[m * Hc + h];
  }
  float pv[8];
#pragma unroll
  for (int v = 0; v < 8; ++v) {
    float s = 0.f;
#pragma unroll
    for (int t = 0; t < 4; ++t) {
      float hv = acc[t][v] + b0v[t];
      hv = hv > 0.f ? hv : 0.01f * hv;
      s += hv * w1v[t];
    }
    pv[v] = s;
  }
#pragma unroll
  for (int off = 1; off < 16; off <<= 1)
#pragma unroll
    for (int v = 0; v < 8; ++v) pv[v] += __shfl_xor(pv[v], off, 32);
  if (lane == 0 || lane == 16) {
    const int roff = (lane == 16) ? 8 : 0;
#pragma unroll
    for (int v = 0; v < 8; ++v) {
      int b = rt * 16 + v + roff;
      pz_mu[(size_t)b * Mc + m] = pv[v] + t_b1[m];
    }
  }
}

// ---------------- decoder: folded GEMM + leaky + layer2 + recon log-lik ----------------
__global__ __launch_bounds__(256) void k_decode(const _Float16* __restrict__ zt16,
                                                const _Float16* __restrict__ W0d,
                                                const float* __restrict__ dec_b0,
                                                const float* __restrict__ dec_W1,
                                                const float* __restrict__ dec_b1,
                                                const float* __restrict__ lv_dec,
                                                const float* __restrict__ y,
                                                float* __restrict__ accum) {
  const int lane = threadIdx.x & 31;
  const int wid  = blockIdx.x * 8 + (threadIdx.x >> 5);   // 40960 waves
  const int rt = wid & 31;
  const int j  = (wid >> 5) & 255;
  const int i  = wid >> 13;
  const _Float16* Ab = zt16 + ((size_t)i * Bc + rt * 16) * 32;
  const _Float16* Wb = W0d + ((size_t)(i * DXc + j)) * Hc * 32;
  const v8f z8 = {0.f,0.f,0.f,0.f,0.f,0.f,0.f,0.f};
  v16h a = ldfrag(Ab, 32, lane);
  v8f acc[4];
#pragma unroll
  for (int t = 0; t < 4; ++t) {
    v16h b = ldfrag(Wb + (size_t)(t * 16) * 32, 32, lane);
    acc[t] = wmma32(a, b, z8);
  }
  const int col = lane & 15;
  float w1v[4], b0v[4];
#pragma unroll
  for (int t = 0; t < 4; ++t) {
    int h = t * 16 + col;
    w1v[t] = dec_W1[j * Hc + h];
    b0v[t] = dec_b0[j * Hc + h];
  }
  float pv[8];
#pragma unroll
  for (int v = 0; v < 8; ++v) {
    float s = 0.f;
#pragma unroll
    for (int t = 0; t < 4; ++t) {
      float hv = acc[t][v] + b0v[t];
      hv = hv > 0.f ? hv : 0.01f * hv;
      s += hv * w1v[t];
    }
    pv[v] = s;
  }
#pragma unroll
  for (int off = 1; off < 16; off <<= 1)
#pragma unroll
    for (int v = 0; v < 8; ++v) pv[v] += __shfl_xor(pv[v], off, 32);
  if (lane == 0 || lane == 16) {
    const int roff = (lane == 16) ? 8 : 0;
    const float lstd = 0.5f * lv_dec[i];
    const float inv  = __expf(-lstd);
    float s = 0.f;
#pragma unroll
    for (int v = 0; v < 8; ++v) {
      int b = rt * 16 + v + roff;
      float px = pv[v] + dec_b1[j];
      float df = (y[((size_t)b * Dc + i) * DXc + j] - px) * inv;
      s += -0.5f * df * df - lstd - 0.5f * kLOG2PI;
    }
    atomicAdd(&accum[0], s);
  }
}

// ---------------- analytic KL ----------------
__global__ __launch_bounds__(256) void k_kl(const float* __restrict__ mu_t,
                                            const float* __restrict__ pz_mu,
                                            const float* __restrict__ lv_enc,
                                            const float* __restrict__ t_lv,
                                            float* __restrict__ accum) {
  __shared__ float red[256];
  int idx = blockIdx.x * 256 + threadIdx.x;   // Bc*Mc = 40960
  int b = idx / Mc, m = idx % Mc, d = m / DZc, zi = m % DZc;
  float lve = lv_enc[d], tl = t_lv[d * DZc + zi];
  float diff = mu_t[((size_t)b * Dc + d) * DZc + zi] - pz_mu[(size_t)b * Mc + m];
  float term = 0.5f * tl - 0.5f * lve + (__expf(lve) + diff * diff) / (2.0f * __expf(tl)) - 0.5f;
  red[threadIdx.x] = term;
  __syncthreads();
  for (int s = 128; s > 0; s >>= 1) {
    if (threadIdx.x < s) red[threadIdx.x] += red[threadIdx.x + s];
    __syncthreads();
  }
  if (threadIdx.x == 0) atomicAdd(&accum[1], red[0]);
}

__global__ void k_final(const float* __restrict__ accum, float* __restrict__ out) {
  if (threadIdx.x == 0 && blockIdx.x == 0)
    out[0] = accum[1] * (1.0f / Bc) - accum[0] * (1.0f / Bc);   // -(recon - kl)
}

// ---------------- launch ----------------
extern "C" void kernel_launch(void* const* d_in, const int* in_sizes, int n_in,
                              void* d_out, int out_size, void* d_ws, size_t ws_size,
                              hipStream_t stream) {
  (void)in_sizes; (void)n_in; (void)out_size; (void)ws_size;
  const float* x        = (const float*)d_in[0];
  const float* y        = (const float*)d_in[1];
  const float* mask_p   = (const float*)d_in[2];
  const float* w_enc    = (const float*)d_in[3];
  const float* w_dec    = (const float*)d_in[4];
  const float* enc_W0   = (const float*)d_in[5];
  const float* enc_b0   = (const float*)d_in[6];
  const float* enc_W1   = (const float*)d_in[7];
  const float* enc_b1   = (const float*)d_in[8];
  const float* dec_W0   = (const float*)d_in[9];
  const float* dec_b0   = (const float*)d_in[10];
  const float* dec_W1   = (const float*)d_in[11];
  const float* dec_b1   = (const float*)d_in[12];
  const float* t_W0     = (const float*)d_in[13];
  const float* t_b0     = (const float*)d_in[14];
  const float* t_W1     = (const float*)d_in[15];
  const float* t_b1     = (const float*)d_in[16];
  const float* lv_enc   = (const float*)d_in[17];
  const float* lv_dec   = (const float*)d_in[18];
  const float* t_lv     = (const float*)d_in[19];

  char* w = (char*)d_ws;
  float*     acc   = (float*)(w + OFF_ACC);
  _Float16*  Aenc  = (_Float16*)(w + OFF_AENC);
  _Float16*  W0e   = (_Float16*)(w + OFF_W0E);
  _Float16*  W0d   = (_Float16*)(w + OFF_W0D);
  _Float16*  tW0h  = (_Float16*)(w + OFF_TW0);
  _Float16*  Atr   = (_Float16*)(w + OFF_ATR);
  float*     zp    = (float*)(w + OFF_ZP);
  float*     mu_t  = (float*)(w + OFF_MUT);
  _Float16*  zt16  = (_Float16*)(w + OFF_ZT);
  float*     pz_mu = (float*)(w + OFF_PZ);

  k_init<<<320, 256, 0, stream>>>(acc, zt16);
  k_build_aenc<<<(Dc * Rc * DXc) / 256, 256, 0, stream>>>(x, y, Aenc);
  k_fold_enc<<<(Dc * DZc * Hc * DXc) / 256, 256, 0, stream>>>(w_enc, enc_W0, W0e);
  k_fold_dec<<<(Dc * DXc * Hc * 32) / 256, 256, 0, stream>>>(w_dec, dec_W0, W0d);
  k_fold_t<<<(Mc * Hc * TMc) / 256, 256, 0, stream>>>(t_W0, tW0h);
  k_encode<<<Dc * DZc * 20, 256, 0, stream>>>(Aenc, W0e, enc_b0, enc_W1,
                                              enc_b1, lv_enc, zp, mu_t, zt16);
  k_build_atr<<<(int)(((long long)Mc * Bc * TMc) / 256), 256, 0, stream>>>(mask_p, zp, Atr);
  k_trans<<<Mc * 4, 256, 0, stream>>>(Atr, tW0h, t_b0, t_W1, t_b1, pz_mu);
  k_decode<<<(Dc * DXc * (Bc / 16)) / 8, 256, 0, stream>>>(zt16, W0d, dec_b0, dec_W1,
                                                           dec_b1, lv_dec, y, acc);
  k_kl<<<(Bc * Mc) / 256, 256, 0, stream>>>(mu_t, pz_mu, lv_enc, t_lv, acc);
  k_final<<<1, 32, 0, stream>>>(acc, (float*)d_out);
}
